// SegmentationGNN_27650999452524
// MI455X (gfx1250) — compile-verified, matrix operand order checked by reading, not run on hardware
//
#include <hip/hip_runtime.h>
#include <hip/hip_bf16.h>

#define HH   192
#define WWID 192
#define HWSZ (HH * WWID)        // 36864
#define BB   4
#define NN   (BB * HWSZ)        // 147456
#define C_IN 32
#define HID  64
#define BN_EPS 1e-5f
#define NTILES (NN / 16)        // 9216 M-tiles
#define GEMM_BLOCKS 288         // 2304 waves -> 4 tiles/wave

// stencil tiling
#define STH 16
#define STW 32
#define SHALO_H (STH + 2)       // 18
#define SHALO_W (STW + 2)       // 34

typedef __attribute__((ext_vector_type(8)))  float  v8f;
typedef __attribute__((ext_vector_type(8)))  __bf16 v8bf;
typedef __attribute__((ext_vector_type(16))) __bf16 v16bf;

// round-to-nearest-even float -> bf16 (bit pattern)
__device__ __forceinline__ unsigned short f2bf(float f) {
  unsigned u = __float_as_uint(f);
  u += 0x7FFFu + ((u >> 16) & 1u);
  return (unsigned short)(u >> 16);
}

// A fragment: two 16B chunks at +0 / +16 elements (ISA 16-bit A 16x32 layout)
__device__ __forceinline__ v16bf load_a16(const unsigned short* p) {
  v8bf lo = *(const v8bf*)(p);
  v8bf hi = *(const v8bf*)(p + 16);
  v16bf r;
#pragma unroll
  for (int i = 0; i < 8; ++i) { r[i] = lo[i]; r[i + 8] = hi[i]; }
  return r;
}

// B fragment: 16 contiguous pre-packed bf16 per lane
__device__ __forceinline__ v16bf load_b16(const unsigned short* p) {
  v8bf lo = *(const v8bf*)(p);
  v8bf hi = *(const v8bf*)(p + 8);
  v16bf r;
#pragma unroll
  for (int i = 0; i < 8; ++i) { r[i] = lo[i]; r[i + 8] = hi[i]; }
  return r;
}

// separable symmetric-norm factors: dinv(h,w) = fh(h)*gw(w)
__device__ __forceinline__ float fh_at(int h) {
  return rsqrtf((float)(1 + (h > 0) + (h < HH - 1)));
}
__device__ __forceinline__ float gw_at(int w) {
  return rsqrtf((float)(1 + (w > 0) + (w < WWID - 1)));
}

// ---------------------------------------------------------------------------
// Pack weights into per-lane B-register order.
//   k = ((p&4)<<2) + ((p&3)<<1) + half*8 + 32*c + odd   (e=2p+odd)
// Pack layout: ((c*nTiles + t)*32 + lane)*16 + e
// ---------------------------------------------------------------------------
__global__ void k_pack(const float* __restrict__ w1,     // [64,32]
                       const float* __restrict__ gcn_w,  // [3,64,64]
                       const float* __restrict__ wf,     // [32,64]
                       unsigned short* __restrict__ w1p,   // 2048
                       unsigned short* __restrict__ gcnp,  // 3*4096
                       unsigned short* __restrict__ wfp) { // 2048
  int idx = blockIdx.x * blockDim.x + threadIdx.x;
  if (idx >= 16384) return;
  if (idx < 2048) {                      // w1: 1 chunk, 4 tiles
    int t = idx >> 9; int r3 = idx & 511; int L = r3 >> 4; int e = r3 & 15;
    int p = e >> 1, odd = e & 1;
    int k = ((p & 4) << 2) + ((p & 3) << 1) + (L >> 4) * 8 + odd;
    w1p[idx] = f2bf(w1[(t * 16 + (L & 15)) * 32 + k]);
  } else if (idx < 2048 + 12288) {       // gcn: 2 chunks, 4 tiles / layer
    int j = idx - 2048;
    int l = j / 4096; int r = j & 4095;
    int c = r >> 11; int r2 = r & 2047;
    int t = r2 >> 9; int r3 = r2 & 511; int L = r3 >> 4; int e = r3 & 15;
    int p = e >> 1, odd = e & 1;
    int k = ((p & 4) << 2) + ((p & 3) << 1) + (L >> 4) * 8 + c * 32 + odd;
    gcnp[j] = f2bf(gcn_w[(l * 64 + t * 16 + (L & 15)) * 64 + k]);
  } else {                               // wf: 2 chunks, 2 tiles
    int j = idx - 2048 - 12288;
    int c = j >> 10; int r2 = j & 1023;
    int t = r2 >> 9; int r3 = r2 & 511; int L = r3 >> 4; int e = r3 & 15;
    int p = e >> 1, odd = e & 1;
    int k = ((p & 4) << 2) + ((p & 3) << 1) + (L >> 4) * 8 + c * 32 + odd;
    wfp[j] = f2bf(wf[(t * 16 + (L & 15)) * 64 + k]);
  }
}

// [B,C,H,W] f32 -> node-major bf16 [N,32]
__global__ void k_to_nodes(const float* __restrict__ feat,
                           unsigned short* __restrict__ Xin) {
  int node = blockIdx.x * blockDim.x + threadIdx.x;
  if (node >= NN) return;
  int b = node / HWSZ; int hw = node - b * HWSZ;
  const float* f = feat + (size_t)b * C_IN * HWSZ + hw;
  unsigned short tmp[C_IN];
#pragma unroll
  for (int c = 0; c < C_IN; ++c) tmp[c] = f2bf(f[(size_t)c * HWSZ]);
  uint4* dst = (uint4*)(Xin + node * C_IN);
#pragma unroll
  for (int q = 0; q < C_IN / 8; ++q) dst[q] = ((uint4*)tmp)[q];
}

// conv1 (K=32) + BN(inference) + ReLU -> Xn bf16 [N,64]; B register-resident
__global__ void __launch_bounds__(256)
k_gemm_conv(const unsigned short* __restrict__ Xin,
            const unsigned short* __restrict__ Wpack,
            const float* __restrict__ b1, const float* __restrict__ gamma,
            const float* __restrict__ beta, const float* __restrict__ rmean,
            const float* __restrict__ rvar,
            unsigned short* __restrict__ Xn) {
  const int lane = threadIdx.x & 31;
  const int waveG = blockIdx.x * (blockDim.x >> 5) + (threadIdx.x >> 5);
  const int nWaves = gridDim.x * (blockDim.x >> 5);
  const int half = lane >> 4, mcol = lane & 15;

  v16bf Bf[4];
#pragma unroll
  for (int t = 0; t < 4; ++t) Bf[t] = load_b16(Wpack + (t * 32 + lane) * 16);

  float sc[4], bt[4], pre[4];  // per-lane epilogue constants (o = t*16+mcol)
#pragma unroll
  for (int t = 0; t < 4; ++t) {
    int o = t * 16 + mcol;
    sc[t] = gamma[o] * rsqrtf(rvar[o] + BN_EPS);
    bt[t] = beta[o];
    pre[t] = b1[o] - rmean[o];
  }

  for (int tile = waveG; tile < NTILES; tile += nWaves) {
    int rowBase = tile * 16;
    const unsigned short* arow = Xin + (rowBase + mcol) * 32 + half * 8;
    v16bf a = load_a16(arow);
    if (tile + nWaves < NTILES)
      __builtin_prefetch((const void*)(arow + nWaves * 16 * 32), 0, 0);
#pragma unroll
    for (int t = 0; t < 4; ++t) {
      v8f acc = {};
      acc = __builtin_amdgcn_wmma_f32_16x16x32_bf16(false, a, false, Bf[t], (short)0, acc, false, false);
      int o = t * 16 + mcol;
#pragma unroll
      for (int r = 0; r < 8; ++r) {
        float y = (acc[r] + pre[t]) * sc[t] + bt[t];
        y = y > 0.f ? y : 0.f;
        Xn[(rowBase + r + half * 8) * 64 + o] = f2bf(y);
      }
    }
  }
}

// xw = Xn @ W^T (K=64) -> f32; B register-resident, grid-stride over M-tiles
__global__ void __launch_bounds__(256)
k_gemm_layer(const unsigned short* __restrict__ Xn,
             const unsigned short* __restrict__ Wpack,
             float* __restrict__ Xw) {
  const int lane = threadIdx.x & 31;
  const int waveG = blockIdx.x * (blockDim.x >> 5) + (threadIdx.x >> 5);
  const int nWaves = gridDim.x * (blockDim.x >> 5);
  const int half = lane >> 4, mcol = lane & 15;

  v16bf Bf[8];
#pragma unroll
  for (int i = 0; i < 8; ++i) Bf[i] = load_b16(Wpack + (i * 32 + lane) * 16);

  for (int tile = waveG; tile < NTILES; tile += nWaves) {
    int rowBase = tile * 16;
    const unsigned short* arow = Xn + (rowBase + mcol) * 64 + half * 8;
    v16bf a0 = load_a16(arow);
    v16bf a1 = load_a16(arow + 32);
    if (tile + nWaves < NTILES)
      __builtin_prefetch((const void*)(arow + nWaves * 16 * 64), 0, 0);

    v8f acc[4];
#pragma unroll
    for (int t = 0; t < 4; ++t) { v8f z = {}; acc[t] = z; }
#pragma unroll
    for (int t = 0; t < 4; ++t) {
      acc[t] = __builtin_amdgcn_wmma_f32_16x16x32_bf16(false, a0, false, Bf[t],     (short)0, acc[t], false, false);
      acc[t] = __builtin_amdgcn_wmma_f32_16x16x32_bf16(false, a1, false, Bf[4 + t], (short)0, acc[t], false, false);
    }
#pragma unroll
    for (int t = 0; t < 4; ++t) {
      int o = t * 16 + mcol;
#pragma unroll
      for (int r = 0; r < 8; ++r) Xw[(rowBase + r + half * 8) * 64 + o] = acc[t][r];
    }
  }
}

// LDS-tiled normalized 3x3 aggregation + bias + ReLU -> Xn bf16.
// Halo tile staged with gfx1250 async global->LDS copies (ASYNCcnt path);
// OOB halo zero-filled so the accumulation needs no bounds checks.
__global__ void __launch_bounds__(256)
k_stencil(const float* __restrict__ Xw, const float* __restrict__ bias,
          unsigned short* __restrict__ Xn) {
  __shared__ float sm[SHALO_H * SHALO_W * 16];   // ~38 KB

  int bid = blockIdx.x;
  int chunk = bid & 3;                 // feature chunk (16 floats)
  int rest = bid >> 2;                 // 0..287
  int wt = rest % (WWID / STW);        // 6 w-tiles
  int ht = (rest / (WWID / STW)) % (HH / STH);  // 12 h-tiles
  int b  = rest / ((WWID / STW) * (HH / STH));  // 4 batches
  int ob = chunk * 16;
  int h0 = ht * STH, w0 = wt * STW;

  // ---- stage halo tile: 16B chunk per lane per iteration ----
  unsigned lds_base = (unsigned)(size_t)(void*)sm;   // low 32 bits = LDS offset
  const int nchunks = SHALO_H * SHALO_W * 4;         // 2448
  for (int i = threadIdx.x; i < nchunks; i += blockDim.x) {
    int q = i & 3;
    int cell = i >> 2;
    int ww = cell % SHALO_W;
    int hh = cell / SHALO_W;
    int nh = h0 - 1 + hh, nw = w0 - 1 + ww;
    unsigned lby = (unsigned)(cell * 64 + q * 16);
    if (nh >= 0 && nh < HH && nw >= 0 && nw < WWID) {
      unsigned long long gp = (unsigned long long)(const void*)
          ((const char*)Xw + (((size_t)(b * HWSZ + nh * WWID + nw)) * 64 + ob) * 4 + q * 16);
      unsigned ld = lds_base + lby;
      asm volatile("global_load_async_to_lds_b128 %0, %1, off"
                   :: "v"(ld), "v"(gp) : "memory");
    } else {
      float4 z = make_float4(0.f, 0.f, 0.f, 0.f);
      *(float4*)((char*)sm + lby) = z;
    }
  }
  asm volatile("s_wait_asynccnt 0x0" ::: "memory");
  __syncthreads();

  // ---- compute: thread -> col = tid&31, rows tid>>5 and +8 ----
  int col = threadIdx.x & (STW - 1);
  int r0 = threadIdx.x >> 5;           // 0..7
  int w = w0 + col;
  float gcol[3];
#pragma unroll
  for (int dw = 0; dw < 3; ++dw) gcol[dw] = gw_at(w + dw - 1);
  float gwc = gw_at(w);

#pragma unroll
  for (int rr = 0; rr < 2; ++rr) {
    int r = r0 + rr * 8;               // 0..15
    int h = h0 + r;
    float acc[16];
#pragma unroll
    for (int i = 0; i < 16; ++i) acc[i] = 0.f;
#pragma unroll
    for (int dh = 0; dh < 3; ++dh) {
      float fr = fh_at(h + dh - 1);
#pragma unroll
      for (int dw = 0; dw < 3; ++dw) {
        float wgt = fr * gcol[dw];
        const float* p = sm + ((r + dh) * SHALO_W + (col + dw)) * 16;
#pragma unroll
        for (int i = 0; i < 16; ++i) acc[i] += wgt * p[i];
      }
    }
    float g0 = fh_at(h) * gwc;
    int node = b * HWSZ + h * WWID + w;
#pragma unroll
    for (int i = 0; i < 16; ++i) {
      float y = acc[i] * g0 + bias[ob + i];
      y = y > 0.f ? y : 0.f;
      Xn[node * 64 + ob + i] = f2bf(y);
    }
  }
}

// out[b,o,h,w] = Xn @ wf^T + bf + features (K=64, 2 col tiles)
__global__ void __launch_bounds__(256)
k_gemm_final(const unsigned short* __restrict__ Xn,
             const unsigned short* __restrict__ Wpack,
             const float* __restrict__ bf,
             const float* __restrict__ feat,
             float* __restrict__ out) {
  const int lane = threadIdx.x & 31;
  const int waveG = blockIdx.x * (blockDim.x >> 5) + (threadIdx.x >> 5);
  const int nWaves = gridDim.x * (blockDim.x >> 5);
  const int half = lane >> 4, mcol = lane & 15;

  v16bf Bf[4];
#pragma unroll
  for (int i = 0; i < 4; ++i) Bf[i] = load_b16(Wpack + (i * 32 + lane) * 16);
  float bo[2] = { bf[mcol], bf[16 + mcol] };

  for (int tile = waveG; tile < NTILES; tile += nWaves) {
    int rowBase = tile * 16;
    const unsigned short* arow = Xn + (rowBase + mcol) * 64 + half * 8;
    v16bf a0 = load_a16(arow);
    v16bf a1 = load_a16(arow + 32);
#pragma unroll
    for (int t = 0; t < 2; ++t) {
      v8f acc = {};
      acc = __builtin_amdgcn_wmma_f32_16x16x32_bf16(false, a0, false, Bf[t],     (short)0, acc, false, false);
      acc = __builtin_amdgcn_wmma_f32_16x16x32_bf16(false, a1, false, Bf[2 + t], (short)0, acc, false, false);
      int o = t * 16 + mcol;
#pragma unroll
      for (int r = 0; r < 8; ++r) {
        int node = rowBase + r + half * 8;
        int b = node / HWSZ; int hw = node - b * HWSZ;
        size_t idx = ((size_t)(b * C_IN + o)) * HWSZ + hw;
        out[idx] = acc[r] + bo[t] + feat[idx];
      }
    }
  }
}

extern "C" void kernel_launch(void* const* d_in, const int* in_sizes, int n_in,
                              void* d_out, int out_size, void* d_ws, size_t ws_size,
                              hipStream_t stream) {
  const float* feat  = (const float*)d_in[0];
  const float* w1    = (const float*)d_in[1];
  const float* b1    = (const float*)d_in[2];
  const float* gamma = (const float*)d_in[3];
  const float* beta  = (const float*)d_in[4];
  const float* rmean = (const float*)d_in[5];
  const float* rvar  = (const float*)d_in[6];
  const float* gcn_w = (const float*)d_in[7];
  const float* gcn_b = (const float*)d_in[8];
  const float* wf    = (const float*)d_in[9];
  const float* bfv   = (const float*)d_in[10];
  // src/dst unused: the 8-connected grid + self-loops is a separable 3x3 stencil.

  char* ws = (char*)d_ws;
  unsigned short* Xin = (unsigned short*)ws;                                      // N*32 bf16
  unsigned short* Xn  = (unsigned short*)(ws + (size_t)NN * 32 * 2);              // N*64 bf16
  float*          Xw  = (float*)(ws + (size_t)NN * 32 * 2 + (size_t)NN * 64 * 2); // N*64 f32
  unsigned short* w1p = (unsigned short*)(ws + (size_t)NN * 32 * 2 +
                                          (size_t)NN * 64 * 2 + (size_t)NN * 64 * 4);
  unsigned short* gcnp = w1p + 2048;
  unsigned short* wfp  = gcnp + 3 * 4096;

  k_pack<<<64, 256, 0, stream>>>(w1, gcn_w, wf, w1p, gcnp, wfp);
  k_to_nodes<<<(NN + 255) / 256, 256, 0, stream>>>(feat, Xin);
  k_gemm_conv<<<GEMM_BLOCKS, 256, 0, stream>>>(Xin, w1p, b1, gamma, beta, rmean, rvar, Xn);
  for (int l = 0; l < 3; ++l) {
    k_gemm_layer<<<GEMM_BLOCKS, 256, 0, stream>>>(Xn, gcnp + l * 4096, Xw);
    k_stencil<<<BB * (HH / STH) * (WWID / STW) * 4, 256, 0, stream>>>(Xw, gcn_b + l * 64, Xn);
  }
  k_gemm_final<<<GEMM_BLOCKS, 256, 0, stream>>>(Xn, wfp, bfv, feat, (float*)d_out);
}
